// HeteroGATLayerReal_46136538693990
// MI455X (gfx1250) — compile-verified, matrix operand order checked by reading, not run on hardware
//
#include <hip/hip_runtime.h>
#include <hip/hip_bf16.h>
#include <stdint.h>

typedef __bf16 bf16_t;
typedef __bf16 v16bf __attribute__((ext_vector_type(16)));
typedef float  v8f   __attribute__((ext_vector_type(8)));

#define IN_DIM 512
#define HD_DIM 256

// ---------------------------------------------------------------------------
// Prepass: f32 -> bf16 (layout-preserving), 4 elems/thread
// ---------------------------------------------------------------------------
__global__ void cvt_f32_bf16(const float* __restrict__ x, bf16_t* __restrict__ y, int n) {
  const int base = (blockIdx.x * blockDim.x + threadIdx.x) * 4;
  if (base >= n) return;
  const float4 f = *(const float4*)(x + base);
  y[base + 0] = (bf16_t)f.x;
  y[base + 1] = (bf16_t)f.y;
  y[base + 2] = (bf16_t)f.z;
  y[base + 3] = (bf16_t)f.w;
}

// Prepass: W[512,256] f32 -> Wt[256][512] bf16 (transpose so B tiles are
// K-contiguous rows, directly async-copyable into LDS)
__global__ void transpose_cvt_bf16(const float* __restrict__ W, bf16_t* __restrict__ Wt) {
  const int i = blockIdx.x * blockDim.x + threadIdx.x;
  if (i >= IN_DIM * HD_DIM) return;
  const int k = i >> 8;      // 0..511
  const int n = i & 255;     // 0..255
  Wt[(size_t)n * IN_DIM + k] = (bf16_t)W[i];
}

// ---------------------------------------------------------------------------
// GEMM: Y[M,256] = Xb[M,512](bf16) @ Wt[256,512](bf16, pre-transposed) + bias
// 256 threads = 8 waves, 128x128 tile, BK=32, double-buffered LDS filled by
// GLOBAL_LOAD_ASYNC_TO_LDS_B128 (ASYNCcnt), fragments via ds_load_b128, WMMA.
// ---------------------------------------------------------------------------
struct Frag { union { uint4 u[2]; v16bf v; }; };

__global__ __launch_bounds__(256)
void gemm_wmma_bf16_async(const bf16_t* __restrict__ Xb, const bf16_t* __restrict__ Wt,
                          const float* __restrict__ bias, float* __restrict__ Y) {
  constexpr int BM = 128, BN = 128, BK = 32, BKP = BK + 8;   // pad keeps 16B align
  constexpr int KSTEPS = IN_DIM / BK;                        // 16
  __shared__ bf16_t As[2][BM * BKP];
  __shared__ bf16_t Bs[2][BN * BKP];

  const int tid  = threadIdx.x;
  const int lane = tid & 31;
  const int wave = tid >> 5;
  const int wr   = wave >> 1;   // 0..3 (M)
  const int wc   = wave & 1;    // 0..1 (N)
  const int lm   = lane & 15;
  const int half = lane >> 4;
  const int mBase = blockIdx.x * BM;
  const int nBase = blockIdx.y * BN;

  // tile-fill coordinates: each thread copies 32B (2 x b128) of A and of B
  const int row = tid >> 1;            // 0..127
  const int c0  = (tid & 1) * 16;      // half-index 0 / 16

  v8f acc[2][4];
#pragma unroll
  for (int mi = 0; mi < 2; ++mi)
#pragma unroll
    for (int ni = 0; ni < 4; ++ni)
      acc[mi][ni] = (v8f){0.f,0.f,0.f,0.f,0.f,0.f,0.f,0.f};

  // async copy of one 128x32 bf16 tile pair into buffer `buf`
  auto issue_tile = [&](int ks, int buf) {
    const unsigned la = (unsigned)(uintptr_t)&As[buf][row * BKP + c0];
    const unsigned lb = (unsigned)(uintptr_t)&Bs[buf][row * BKP + c0];
    const bf16_t* ga = Xb + (size_t)(mBase + row) * IN_DIM + ks * BK + c0;
    const bf16_t* gb = Wt + (size_t)(nBase + row) * IN_DIM + ks * BK + c0;
    // inst offset applies to BOTH lds and global address (ISA 15.18.3)
    asm volatile(
        "global_load_async_to_lds_b128 %0, %2, off\n\t"
        "global_load_async_to_lds_b128 %0, %2, off offset:16\n\t"
        "global_load_async_to_lds_b128 %1, %3, off\n\t"
        "global_load_async_to_lds_b128 %1, %3, off offset:16"
        :: "v"(la), "v"(lb), "v"(ga), "v"(gb) : "memory");
  };

  issue_tile(0, 0);

  for (int ks = 0; ks < KSTEPS; ++ks) {
    const int cur = ks & 1;
    if (ks + 1 < KSTEPS) {
      issue_tile(ks + 1, cur ^ 1);                       // prefetch next tile
      asm volatile("s_wait_asynccnt 4" ::: "memory");    // current tile landed
    } else {
      asm volatile("s_wait_asynccnt 0" ::: "memory");
    }
    __syncthreads();

    // ISA 16-bit A/B layout: lanes 0-15 hold K 0-7 & 16-23, lanes 16-31 K 8-15 & 24-31
    Frag a[2], b[4];
#pragma unroll
    for (int mi = 0; mi < 2; ++mi) {
      const int m = wr * 32 + mi * 16 + lm;
      a[mi].u[0] = *(const uint4*)&As[cur][m * BKP + half * 8];
      a[mi].u[1] = *(const uint4*)&As[cur][m * BKP + 16 + half * 8];
    }
#pragma unroll
    for (int ni = 0; ni < 4; ++ni) {
      const int n = wc * 64 + ni * 16 + lm;
      b[ni].u[0] = *(const uint4*)&Bs[cur][n * BKP + half * 8];
      b[ni].u[1] = *(const uint4*)&Bs[cur][n * BKP + 16 + half * 8];
    }
#pragma unroll
    for (int mi = 0; mi < 2; ++mi)
#pragma unroll
      for (int ni = 0; ni < 4; ++ni)
        acc[mi][ni] = __builtin_amdgcn_wmma_f32_16x16x32_bf16(
            false, a[mi].v, false, b[ni].v, (short)0, acc[mi][ni], false, false);
    __syncthreads();   // all reads of `cur` done before it is refilled
  }

  // D layout: VGPR r -> M = half*8 + r, N = lane&15
#pragma unroll
  for (int mi = 0; mi < 2; ++mi) {
#pragma unroll
    for (int ni = 0; ni < 4; ++ni) {
      const int n = nBase + wc * 64 + ni * 16 + lm;
      const float bv = bias[n];
#pragma unroll
      for (int r = 0; r < 8; ++r) {
        const int m = mBase + wr * 32 + mi * 16 + half * 8 + r;
        Y[(size_t)m * HD_DIM + n] = acc[mi][ni][r] + bv;
      }
    }
  }
}

// ---------------------------------------------------------------------------
// attn[n,h] = sum_d Wh[n,h,d] * att[h,d]
// ---------------------------------------------------------------------------
__global__ void attn_proj(const float* __restrict__ Wh, const float* __restrict__ att,
                          float* __restrict__ out, int NH8) {
  int i = blockIdx.x * blockDim.x + threadIdx.x;
  if (i >= NH8) return;
  const int h = i & 7;
  const float* w = Wh + (size_t)i * 32;   // (n*8+h)*32 == n*256 + h*32
  const float* a = att + h * 32;
  float s = 0.f;
#pragma unroll
  for (int d = 0; d < 32; ++d) s += w[d] * a[d];
  out[i] = s;
}

// order-preserving float<->uint for atomicMax-based segment max
__device__ __forceinline__ unsigned f2ord(float f) {
  int i = __float_as_int(f);
  return (i < 0) ? ~(unsigned)i : ((unsigned)i | 0x80000000u);
}
__device__ __forceinline__ float ord2f(unsigned u) {
  int i = (u & 0x80000000u) ? (int)(u & 0x7fffffffu) : ~(int)u;
  return __int_as_float(i);
}

__global__ void edge_score_max(const float* __restrict__ as, const float* __restrict__ ad,
                               const int* __restrict__ src, const int* __restrict__ dst,
                               float* __restrict__ e, unsigned* __restrict__ menc, int EH) {
  int i = blockIdx.x * blockDim.x + threadIdx.x;
  if (i >= EH) return;
  const int ed = i >> 3, h = i & 7;
  float v = as[src[ed] * 8 + h] + ad[dst[ed] * 8 + h];
  v = v > 0.f ? v : 0.2f * v;                       // leaky_relu(0.2)
  e[i] = v;
  atomicMax(&menc[dst[ed] * 8 + h], f2ord(v));
}

__global__ void edge_exp_sum(float* __restrict__ e, const int* __restrict__ dst,
                             const unsigned* __restrict__ menc, float* __restrict__ z, int EH) {
  int i = blockIdx.x * blockDim.x + threadIdx.x;
  if (i >= EH) return;
  const int ed = i >> 3, h = i & 7;
  const float m = ord2f(menc[dst[ed] * 8 + h]);
  const float ex = __expf(e[i] - m);
  e[i] = ex;
  atomicAdd(&z[dst[ed] * 8 + h], ex);
}

// 4 edges per block, 64 lanes per edge, float4 gather + 4 f32 atomic adds.
__global__ __launch_bounds__(256)
void edge_aggregate(const float* __restrict__ Wh, const float* __restrict__ e,
                    const float* __restrict__ z,
                    const int* __restrict__ src, const int* __restrict__ dst,
                    float* __restrict__ out) {
  const int ed = blockIdx.x * 4 + (threadIdx.x >> 6);
  const int q  = threadIdx.x & 63;        // float4 index within the 256-wide row
  const int h  = q >> 3;
  const int s  = src[ed], d = dst[ed];
  const float a = e[(size_t)ed * 8 + h] / z[d * 8 + h];
  const float4 w = *(const float4*)(Wh + (size_t)s * 256 + q * 4);
  float* o = out + (size_t)d * 256 + q * 4;
  atomicAdd(o + 0, w.x * a);
  atomicAdd(o + 1, w.y * a);
  atomicAdd(o + 2, w.z * a);
  atomicAdd(o + 3, w.w * a);
}

__global__ void relu_inplace(float* __restrict__ p, int n) {
  int i = blockIdx.x * blockDim.x + threadIdx.x;
  if (i < n) p[i] = fmaxf(p[i], 0.f);
}

// ---------------------------------------------------------------------------
extern "C" void kernel_launch(void* const* d_in, const int* in_sizes, int n_in,
                              void* d_out, int out_size, void* d_ws, size_t ws_size,
                              hipStream_t stream) {
  const int NP = 16384, NA = 16384, NS = 256;
  const int E = 131072, ES = 65536;

  const float* xP = (const float*)d_in[0];
  const float* xA = (const float*)d_in[1];
  const float* xS = (const float*)d_in[2];
  const float* WP   = (const float*)d_in[3];  const float* bP   = (const float*)d_in[4];
  const float* WA   = (const float*)d_in[5];  const float* bA   = (const float*)d_in[6];
  const float* Wp2p = (const float*)d_in[7];  const float* bp2p = (const float*)d_in[8];
  const float* Wp2a = (const float*)d_in[9];  const float* bp2a = (const float*)d_in[10];
  const float* Wa2p = (const float*)d_in[11]; const float* ba2p = (const float*)d_in[12];
  const float* Wa2a = (const float*)d_in[13]; const float* ba2a = (const float*)d_in[14];
  const float* Wp2s = (const float*)d_in[15]; const float* bp2s = (const float*)d_in[16];
  const float* Wa2s = (const float*)d_in[17]; const float* ba2s = (const float*)d_in[18];
  const float* Win  = (const float*)d_in[19]; const float* bin  = (const float*)d_in[20];
  const float* att_p2p_s = (const float*)d_in[21]; const float* att_p2p_d = (const float*)d_in[22];
  const float* att_p2a_s = (const float*)d_in[23]; const float* att_p2a_d = (const float*)d_in[24];
  const float* att_a2p_s = (const float*)d_in[25]; const float* att_a2p_d = (const float*)d_in[26];
  const float* att_a2a_s = (const float*)d_in[27]; const float* att_a2a_d = (const float*)d_in[28];
  const float* att_p2s_s = (const float*)d_in[29]; const float* att_p2s_d = (const float*)d_in[30];
  const float* att_a2s_s = (const float*)d_in[31]; const float* att_a2s_d = (const float*)d_in[32];
  const int* src_p2p = (const int*)d_in[33]; const int* dst_p2p = (const int*)d_in[34];
  const int* src_p2a = (const int*)d_in[35]; const int* dst_p2a = (const int*)d_in[36];
  const int* src_a2p = (const int*)d_in[37]; const int* dst_a2p = (const int*)d_in[38];
  const int* src_a2a = (const int*)d_in[39]; const int* dst_a2a = (const int*)d_in[40];
  const int* src_p2s = (const int*)d_in[41]; const int* dst_p2s = (const int*)d_in[42];
  const int* src_a2s = (const int*)d_in[43]; const int* dst_a2s = (const int*)d_in[44];

  float* outF = (float*)d_out;
  float* hP = outF;                                  // [NP,256] starts as Whp
  float* hA = outF + (size_t)NP * 256;               // [NA,256] starts as Wha
  float* hS = outF + (size_t)(NP + NA) * 256;        // [NS,256] starts as Whin

  float* ws = (float*)d_ws;
  size_t off = 0;
  auto take = [&](size_t nFloats) { float* p = ws + off; off += nFloats; return p; };

  // bf16 staging (sizes in float units = halfs/2)
  bf16_t* xPb = (bf16_t*)take((size_t)NP * IN_DIM / 2);
  bf16_t* xAb = (bf16_t*)take((size_t)NA * IN_DIM / 2);
  bf16_t* xSb = (bf16_t*)take((size_t)NS * IN_DIM / 2);
  bf16_t* Wt[9];
  for (int i = 0; i < 9; ++i) Wt[i] = (bf16_t*)take((size_t)IN_DIM * HD_DIM / 2);

  float* Whp2p = take((size_t)NP * 256);
  float* Whp2a = take((size_t)NP * 256);
  float* Wha2p = take((size_t)NA * 256);
  float* Wha2a = take((size_t)NA * 256);
  float* Whp2s = take((size_t)NP * 256);
  float* Wha2s = take((size_t)NA * 256);

  float* s_p2p = take((size_t)NP * 8); float* s_p2a = take((size_t)NP * 8);
  float* s_a2p = take((size_t)NA * 8); float* s_a2a = take((size_t)NA * 8);
  float* s_p2s = take((size_t)NP * 8); float* s_a2s = take((size_t)NA * 8);
  float* dv_p2p = take((size_t)NP * 8); float* dv_p2a = take((size_t)NA * 8);
  float* dv_a2p = take((size_t)NP * 8); float* dv_a2a = take((size_t)NA * 8);
  float* dv_p2s = take((size_t)NS * 8); float* dv_a2s = take((size_t)NS * 8);

  const size_t mzStart = off;          // menc + z zeroed together each call
  unsigned* m_p2p = (unsigned*)take((size_t)NP * 8);
  unsigned* m_p2a = (unsigned*)take((size_t)NA * 8);
  unsigned* m_a2p = (unsigned*)take((size_t)NP * 8);
  unsigned* m_a2a = (unsigned*)take((size_t)NA * 8);
  unsigned* m_p2s = (unsigned*)take((size_t)NS * 8);
  unsigned* m_a2s = (unsigned*)take((size_t)NS * 8);
  float* z_p2p = take((size_t)NP * 8); float* z_p2a = take((size_t)NA * 8);
  float* z_a2p = take((size_t)NP * 8); float* z_a2a = take((size_t)NA * 8);
  float* z_p2s = take((size_t)NS * 8); float* z_a2s = take((size_t)NS * 8);
  const size_t mzBytes = (off - mzStart) * sizeof(float);

  float* e_p2p = take((size_t)E * 8);  float* e_p2a = take((size_t)E * 8);
  float* e_a2p = take((size_t)E * 8);  float* e_a2a = take((size_t)E * 8);
  float* e_p2s = take((size_t)ES * 8); float* e_a2s = take((size_t)ES * 8);

  hipMemsetAsync(ws + mzStart, 0, mzBytes, stream);

  // ---- precision staging: activations to bf16 once, weights to bf16^T ----
  auto cvt = [&](const float* x, bf16_t* y, size_t n) {
    cvt_f32_bf16<<<(unsigned)((n / 4 + 255) / 256), 256, 0, stream>>>(x, y, (int)n);
  };
  cvt(xP, xPb, (size_t)NP * IN_DIM);
  cvt(xA, xAb, (size_t)NA * IN_DIM);
  cvt(xS, xSb, (size_t)NS * IN_DIM);
  const float* Wsrc[9] = {WP, WA, Win, Wp2p, Wp2a, Wa2p, Wa2a, Wp2s, Wa2s};
  for (int i = 0; i < 9; ++i)
    transpose_cvt_bf16<<<(IN_DIM * HD_DIM + 255) / 256, 256, 0, stream>>>(Wsrc[i], Wt[i]);

  // ---- 9 WMMA GEMMs (self transforms land directly in d_out) ----
  dim3 blk(256);
  dim3 gP(NP / 128, 2), gA(NA / 128, 2), gS(NS / 128, 2);
  gemm_wmma_bf16_async<<<gP, blk, 0, stream>>>(xPb, Wt[0], bP,   hP);
  gemm_wmma_bf16_async<<<gA, blk, 0, stream>>>(xAb, Wt[1], bA,   hA);
  gemm_wmma_bf16_async<<<gS, blk, 0, stream>>>(xSb, Wt[2], bin,  hS);
  gemm_wmma_bf16_async<<<gP, blk, 0, stream>>>(xPb, Wt[3], bp2p, Whp2p);
  gemm_wmma_bf16_async<<<gP, blk, 0, stream>>>(xPb, Wt[4], bp2a, Whp2a);
  gemm_wmma_bf16_async<<<gA, blk, 0, stream>>>(xAb, Wt[5], ba2p, Wha2p);
  gemm_wmma_bf16_async<<<gA, blk, 0, stream>>>(xAb, Wt[6], ba2a, Wha2a);
  gemm_wmma_bf16_async<<<gP, blk, 0, stream>>>(xPb, Wt[7], bp2s, Whp2s);
  gemm_wmma_bf16_async<<<gA, blk, 0, stream>>>(xAb, Wt[8], ba2s, Wha2s);

  // ---- attention projections ----
  auto attn = [&](const float* Wh, const float* av, float* o, int N) {
    const int t = N * 8;
    attn_proj<<<(t + 255) / 256, 256, 0, stream>>>(Wh, av, o, t);
  };
  attn(Whp2p, att_p2p_s, s_p2p, NP);  attn(hP, att_p2p_d, dv_p2p, NP);
  attn(Whp2a, att_p2a_s, s_p2a, NP);  attn(hA, att_p2a_d, dv_p2a, NA);
  attn(Wha2p, att_a2p_s, s_a2p, NA);  attn(hP, att_a2p_d, dv_a2p, NP);
  attn(Wha2a, att_a2a_s, s_a2a, NA);  attn(hA, att_a2a_d, dv_a2a, NA);
  attn(Whp2s, att_p2s_s, s_p2s, NP);  attn(hS, att_p2s_d, dv_p2s, NS);
  attn(Wha2s, att_a2s_s, s_a2s, NA);  attn(hS, att_a2s_d, dv_a2s, NS);

  // ---- per-relation edge softmax + scatter aggregate ----
  auto rel = [&](const float* Wh, const float* sv, const float* dv,
                 const int* src, const int* dst, float* ebuf, unsigned* menc,
                 float* z, float* out, int Ecnt) {
    const int eh = Ecnt * 8;
    edge_score_max<<<(eh + 255) / 256, 256, 0, stream>>>(sv, dv, src, dst, ebuf, menc, eh);
    edge_exp_sum  <<<(eh + 255) / 256, 256, 0, stream>>>(ebuf, dst, menc, z, eh);
    edge_aggregate<<<Ecnt / 4, 256, 0, stream>>>(Wh, ebuf, z, src, dst, out);
  };
  rel(Whp2p, s_p2p, dv_p2p, src_p2p, dst_p2p, e_p2p, m_p2p, z_p2p, hP, E);
  rel(Whp2a, s_p2a, dv_p2a, src_p2a, dst_p2a, e_p2a, m_p2a, z_p2a, hA, E);
  rel(Wha2p, s_a2p, dv_a2p, src_a2p, dst_a2p, e_a2p, m_a2p, z_a2p, hP, E);
  rel(Wha2a, s_a2a, dv_a2a, src_a2a, dst_a2a, e_a2a, m_a2a, z_a2a, hA, E);
  rel(Whp2s, s_p2s, dv_p2s, src_p2s, dst_p2s, e_p2s, m_p2s, z_p2s, hS, ES);
  rel(Wha2s, s_a2s, dv_a2s, src_a2s, dst_a2s, e_a2s, m_a2s, z_a2s, hS, ES);

  relu_inplace<<<(out_size + 255) / 256, 256, 0, stream>>>(outF, out_size);
}
